// DynamicConv2D_79680233275700
// MI455X (gfx1250) — matrix operand order repaired and checked by reading, not compiled, not run on hardware
//
#include <hip/hip_runtime.h>
#include <hip/hip_bf16.h>

typedef _Float16 v16h __attribute__((ext_vector_type(16)));
typedef _Float16 v8h  __attribute__((ext_vector_type(8)));
typedef float    v8f  __attribute__((ext_vector_type(8)));
typedef float    v4f  __attribute__((ext_vector_type(4)));
typedef int      v4i  __attribute__((ext_vector_type(4)));

#define B_   32
#define H_   56
#define W_   56
#define C_   256
#define F_   256
#define KD   2304        // 3*3*256  (GEMM K dimension)
#define P_   3136        // 56*56    (GEMM M dimension per sample)
#define NK   4
#define HID  64
#define TEMP 30.0f
#define NSTEP 72         // KD / 32

#define A_STRIDE 40      // 32 + 8 halfs skew  (16B-aligned rows, conflict-free)
#define B_STRIDE 264     // 256 + 8 halfs skew (16B-aligned rows)

// Probe for the gfx1250 async global->LDS path (ASYNCcnt).
#if defined(__has_builtin)
#  if __has_builtin(__builtin_amdgcn_global_load_async_to_lds_b128)
#    define USE_ASYNC_LDS 1
#  endif
#endif
#ifndef USE_ASYNC_LDS
#  define USE_ASYNC_LDS 0
#endif

#if defined(__has_builtin)
#  if __has_builtin(__builtin_amdgcn_s_wait_asynccnt)
#    define HAVE_WAIT_ASYNC 1
#  endif
#endif
#ifndef HAVE_WAIT_ASYNC
#  define HAVE_WAIT_ASYNC 0
#endif

// ---------------------------------------------------------------------------
// Kernel 1: attention router.  grid = (32), block = 256
// ---------------------------------------------------------------------------
__global__ void __launch_bounds__(256)
dynconv_attn_kernel(const float* __restrict__ x,
                    const float* __restrict__ w1, const float* __restrict__ b1,
                    const float* __restrict__ w2, const float* __restrict__ b2,
                    float* __restrict__ attn) {
    __shared__ float pooled[C_];
    __shared__ float hbuf[HID];
    __shared__ float lg[NK];

    const int b = blockIdx.x;
    const int c = threadIdx.x;                    // one channel per thread
    const float* xb = x + (size_t)b * P_ * C_ + c;

    float s0 = 0.f, s1 = 0.f, s2 = 0.f, s3 = 0.f;
    for (int p = 0; p < P_; p += 4) {             // coalesced: lanes -> channels
        s0 += xb[(p + 0) * C_];
        s1 += xb[(p + 1) * C_];
        s2 += xb[(p + 2) * C_];
        s3 += xb[(p + 3) * C_];
    }
    pooled[c] = (s0 + s1 + s2 + s3) * (1.0f / (float)P_);
    __syncthreads();

    if (c < HID) {
        float acc = b1[c];
        for (int i = 0; i < C_; ++i) acc += pooled[i] * w1[i * HID + c];
        hbuf[c] = acc > 0.f ? acc : 0.f;
    }
    __syncthreads();

    if (c < NK) {
        float acc = b2[c];
        for (int j = 0; j < HID; ++j) acc += hbuf[j] * w2[j * NK + c];
        lg[c] = acc;
    }
    __syncthreads();

    if (c == 0) {
        float m = lg[0];
        for (int k = 1; k < NK; ++k) m = fmaxf(m, lg[k]);
        float e[NK], s = 0.f;
        for (int k = 0; k < NK; ++k) { e[k] = __expf((lg[k] - m) * (1.0f / TEMP)); s += e[k]; }
        float inv = 1.0f / s;
        for (int k = 0; k < NK; ++k) attn[b * NK + k] = e[k] * inv;
    }
}

// ---------------------------------------------------------------------------
// Kernel 2: per-sample kernel aggregation (linearity of conv).
// grid = (KD, B_), block = 256.  Writes f16 weights packed [b][kd][f].
// ---------------------------------------------------------------------------
__global__ void __launch_bounds__(256)
dynconv_aggw_kernel(const float* __restrict__ kernels,
                    const float* __restrict__ biases,
                    const float* __restrict__ attn,
                    _Float16* __restrict__ wf16,
                    float* __restrict__ bias_agg) {
    const int f  = threadIdx.x;
    const int kk = blockIdx.x;
    const int b  = blockIdx.y;
    const float a0 = attn[b * NK + 0];
    const float a1 = attn[b * NK + 1];
    const float a2 = attn[b * NK + 2];
    const float a3 = attn[b * NK + 3];
    const int KSZ = KD * F_;                      // 589824 elems per candidate kernel
    const int idx = kk * F_ + f;
    float w = a0 * kernels[idx]
            + a1 * kernels[KSZ + idx]
            + a2 * kernels[2 * KSZ + idx]
            + a3 * kernels[3 * KSZ + idx];
    wf16[(size_t)b * KSZ + idx] = (_Float16)w;
    if (kk == 0) {
        bias_agg[b * F_ + f] = a0 * biases[f] + a1 * biases[F_ + f]
                             + a2 * biases[2 * F_ + f] + a3 * biases[3 * F_ + f];
    }
}

// ---------------------------------------------------------------------------
// Kernel 3: implicit-GEMM conv, double-buffered LDS, register-prefetched A,
// async(ASYNCcnt)-staged B when the toolchain exposes the builtin.
// grid = (49, 32), block = 256 (8 waves: 4(M) x 2(N)), K-step 32.
// ---------------------------------------------------------------------------
__global__ void __launch_bounds__(256)
dynconv_gemm_kernel(const float* __restrict__ x,
                    const _Float16* __restrict__ wf16,
                    const float* __restrict__ bias_agg,
                    float* __restrict__ out) {
    __shared__ __align__(16) _Float16 As[2][64 * A_STRIDE];
    __shared__ __align__(16) _Float16 Bs[2][32 * B_STRIDE];

    const int tile = blockIdx.x;                  // 0..48
    const int b    = blockIdx.y;                  // sample
    const int tid  = threadIdx.x;
    const int lane = tid & 31;
    const int wv   = tid >> 5;                    // wave 0..7
    const int wm   = wv >> 1;                     // 0..3  (M sub-tile)
    const int wn   = wv & 1;                      // 0..1  (N half)
    const int p0   = tile * 64;

    const float*    xb = x    + (size_t)b * P_ * C_;
    const _Float16* wb = wf16 + (size_t)b * KD * F_;

    v8f acc[8];
#pragma unroll
    for (int j = 0; j < 8; ++j) acc[j] = (v8f){};

    // A-tile loader mapping: 64 rows x 32 halfs, 4 threads/row, 8 halfs each
    const int ar = tid >> 2;
    const int ac = (tid & 3) * 8;
    const int p  = p0 + ar;
    const int ph = p / W_;
    const int pw = p % W_;
    // per-thread invariant pieces of the implicit-gemm gather
    const int base0 = ((ph - 1) * W_ + (pw - 1)) * C_ + ac;
    unsigned vmask = 0;
#pragma unroll
    for (int s = 0; s < 9; ++s) {                 // compile-time kh/kw
        const int kh = s / 3, kw = s % 3;
        const int hh = ph + kh - 1, ww = pw + kw - 1;
        vmask |= ((hh >= 0 && hh < H_ && ww >= 0 && ww < W_) ? 1u : 0u) << s;
    }

    // B-tile loader mapping: 32 rows x 256 halfs, 8 threads/row, 32 halfs each
    const int br = tid >> 3;
    const int bc = (tid & 7) * 32;

    const int m_lane = lane & 15;
    const int grp    = lane >> 4;

    // ---- stage helpers -----------------------------------------------------
    auto fetchA = [&](int i, v4f& f0, v4f& f1) {
        const int s  = i >> 3;                    // tap 0..8 (uniform)
        const int c0 = (i & 7) << 5;              // channel base (uniform)
        const int kh = (s * 11) >> 5;             // s/3 for s in [0,8]
        const int kw = s - kh * 3;
        const int koff = (kh * W_ + kw) * C_;     // uniform tap offset
        f0 = (v4f){};
        f1 = (v4f){};
        if ((vmask >> s) & 1u) {
            const float* src = xb + (base0 + koff + c0);
            f0 = *(const v4f*)src;
            f1 = *(const v4f*)(src + 4);
        }
    };
    auto storeA = [&](int buf, v4f f0, v4f f1) {
        v8h av;
        av[0] = (_Float16)f0.x; av[1] = (_Float16)f0.y;
        av[2] = (_Float16)f0.z; av[3] = (_Float16)f0.w;
        av[4] = (_Float16)f1.x; av[5] = (_Float16)f1.y;
        av[6] = (_Float16)f1.z; av[7] = (_Float16)f1.w;
        *(v8h*)&As[buf][ar * A_STRIDE + ac] = av;
    };

#if USE_ASYNC_LDS
    auto issueB = [&](int i, int buf) {
        const int k0 = i << 5;
        const _Float16* src = wb + (size_t)(k0 + br) * F_ + bc;    // 64 bytes/thread
        auto gp = (__attribute__((address_space(1))) v4i*)src;
        auto lp = (__attribute__((address_space(3))) v4i*)&Bs[buf][br * B_STRIDE + bc];
#pragma unroll
        for (int q = 0; q < 4; ++q)
            __builtin_amdgcn_global_load_async_to_lds_b128(gp + q, lp + q, 0, 0);
    };
    auto waitB = [&]() {
#if HAVE_WAIT_ASYNC
        __builtin_amdgcn_s_wait_asynccnt(0);
#endif
    };
#else
    v8h bregs[4];
    auto fetchB = [&](int i) {
        const int k0 = i << 5;
        const _Float16* src = wb + (size_t)(k0 + br) * F_ + bc;
#pragma unroll
        for (int q = 0; q < 4; ++q) bregs[q] = *(const v8h*)(src + q * 8);
    };
    auto storeB = [&](int buf) {
        _Float16* dst = &Bs[buf][br * B_STRIDE + bc];
#pragma unroll
        for (int q = 0; q < 4; ++q) *(v8h*)(dst + q * 8) = bregs[q];
    };
#endif

    // ---- prologue: stage step 0 into buffer 0 ------------------------------
    v4f a0f, a1f;
    fetchA(0, a0f, a1f);
#if USE_ASYNC_LDS
    issueB(0, 0);
#else
    fetchB(0);
#endif
    storeA(0, a0f, a1f);
#if USE_ASYNC_LDS
    waitB();
#else
    storeB(0);
#endif
    __syncthreads();

    // ---- main pipelined loop ----------------------------------------------
    for (int i = 0; i < NSTEP; ++i) {
        const int cur = i & 1;
        const int nxt = cur ^ 1;
        const bool more = (i + 1) < NSTEP;

        // prefetch step i+1 (global loads in flight during WMMA block)
        if (more) {
            fetchA(i + 1, a0f, a1f);
#if USE_ASYNC_LDS
            issueB(i + 1, nxt);
#else
            fetchB(i + 1);
#endif
        }

        // A fragment (16x32 f16): lanes 0-15 K 0..7/16..23, lanes 16-31 K 8..15/24..31
        const _Float16* arow = &As[cur][(wm * 16 + m_lane) * A_STRIDE];
        v8h alo = *(const v8h*)(arow + grp * 8);
        v8h ahi = *(const v8h*)(arow + 16 + grp * 8);
        v16h afrag;
#pragma unroll
        for (int q = 0; q < 8; ++q) { afrag[q] = alo[q]; afrag[8 + q] = ahi[q]; }

        // 8 WMMAs: B fragment lane = N column (lane&15), K in same order as A
#pragma unroll
        for (int j = 0; j < 8; ++j) {
            const int n0 = wn * 128 + j * 16;
            v16h bfrag;
#pragma unroll
            for (int q = 0; q < 8; ++q) {
                bfrag[q]     = Bs[cur][(grp * 8 + q) * B_STRIDE + n0 + m_lane];
                bfrag[8 + q] = Bs[cur][(16 + grp * 8 + q) * B_STRIDE + n0 + m_lane];
            }
            acc[j] = __builtin_amdgcn_wmma_f32_16x16x32_f16(
                false, afrag, false, bfrag, (short)0, acc[j], false, false);
        }

        // land step i+1 into the other buffer
        if (more) {
            storeA(nxt, a0f, a1f);
#if USE_ASYNC_LDS
            waitB();
#else
            storeB(nxt);
#endif
        }
        __syncthreads();
    }

    // ---- epilogue: bias + store.  C/D layout: M = r + 8*(lane>>4), N = lane&15
    const int m_base = wm * 16 + (lane >> 4) * 8;
    const int nlo    = lane & 15;
    float* outb = out + (size_t)b * P_ * F_;
#pragma unroll
    for (int j = 0; j < 8; ++j) {
        const int n = wn * 128 + j * 16 + nlo;
        const float bias = bias_agg[b * F_ + n];
#pragma unroll
        for (int r = 0; r < 8; ++r) {
            const int mm = m_base + r;
            outb[(size_t)(p0 + mm) * F_ + n] = acc[j][r] + bias;
        }
    }
}

// ---------------------------------------------------------------------------
extern "C" void kernel_launch(void* const* d_in, const int* in_sizes, int n_in,
                              void* d_out, int out_size, void* d_ws, size_t ws_size,
                              hipStream_t stream) {
    const float* x       = (const float*)d_in[0];
    const float* kernels = (const float*)d_in[1];
    const float* biases  = (const float*)d_in[2];
    const float* w1      = (const float*)d_in[3];
    const float* b1      = (const float*)d_in[4];
    const float* w2      = (const float*)d_in[5];
    const float* b2      = (const float*)d_in[6];
    float* out = (float*)d_out;

    // workspace layout
    float*    attn     = (float*)d_ws;                    // 32*4 floats
    float*    bias_agg = attn + B_ * NK;                  // 32*256 floats
    _Float16* wf16     = (_Float16*)(bias_agg + B_ * F_); // 32*2304*256 halfs (~37.7 MB)

    dynconv_attn_kernel<<<B_, 256, 0, stream>>>(x, w1, b1, w2, b2, attn);

    dynconv_aggw_kernel<<<dim3(KD, B_), 256, 0, stream>>>(kernels, biases, attn,
                                                          wf16, bias_agg);

    dynconv_gemm_kernel<<<dim3(P_ / 64, B_), 256, 0, stream>>>(x, wf16, bias_agg, out);
}
